// pGNNNet_33603824124481
// MI455X (gfx1250) — matrix operand order repaired and checked by reading, not compiled, not run on hardware
//
#include <hip/hip_runtime.h>
#include <math.h>

// ---------------------------------------------------------------------------
// pGNN pipeline for gfx1250 (MI455X, wave32, WMMA)
//   f = relu(x@W1+b1)@W2+b2          -- fused WMMA GEMM (f16 in, f32 acc)
//   K=2 rounds of pGNN propagation   -- wave-per-edge, lane-per-channel
//   log_softmax over 32 channels     -- wave-per-row
// ---------------------------------------------------------------------------

typedef _Float16 half_t;
typedef half_t v16h __attribute__((ext_vector_type(16)));
typedef float  v8f  __attribute__((ext_vector_type(8)));

union V16H { v16h v; half_t h[16]; };
union V8F  { v8f  v; float  f[8];  };

#define N_IN   512
#define N_HID  64
#define N_OUT  32
#define MU_C   0.1f
#define PEXP_C (-0.25f)   // 0.5*(P-2) with P=1.5
#define EPS_C  1e-8f

// ---------------- prep: convert W1/W2 (f32 row-major) to B-fragment f16 -----
// B-fragment layout (16-bit B, 32xN tile): linear index
//   idx = (((kblk*NT + nt)*32) + lane)*16 + e,  K = kblk*32 + e + (lane>>4)*16,
//   n  = nt*16 + (lane&15)
__global__ void convert_w1_kernel(const float* __restrict__ W1, half_t* __restrict__ W1f) {
  int tid = blockIdx.x * blockDim.x + threadIdx.x;
  if (tid >= N_IN * N_HID) return;
  int e    = tid & 15;
  int lane = (tid >> 4) & 31;
  int nt   = (tid >> 9) & 3;
  int kb   = tid >> 11;
  int k    = kb * 32 + e + (lane >> 4) * 16;
  int col  = nt * 16 + (lane & 15);
  W1f[tid] = (half_t)W1[k * N_HID + col];
}

__global__ void convert_w2_kernel(const float* __restrict__ W2, half_t* __restrict__ W2f) {
  int tid = blockIdx.x * blockDim.x + threadIdx.x;
  if (tid >= N_HID * N_OUT) return;
  int e    = tid & 15;
  int lane = (tid >> 4) & 31;
  int nt   = (tid >> 9) & 1;
  int ks   = tid >> 10;
  int k    = ks * 32 + e + (lane >> 4) * 16;
  int col  = nt * 16 + (lane & 15);
  W2f[tid] = (half_t)W2[k * N_OUT + col];
}

// ---------------- utility ---------------------------------------------------
__global__ void zero_kernel(float* __restrict__ p, long long cnt) {
  long long i = (long long)blockIdx.x * blockDim.x + threadIdx.x;
  if (i < cnt) p[i] = 0.0f;
}

__global__ void deg_kernel(const int* __restrict__ src, const int* __restrict__ dst,
                           float* __restrict__ deg, int n_edges) {
  long long i = (long long)blockIdx.x * blockDim.x + threadIdx.x;
  if (i < n_edges) {
    atomicAdd(&deg[src[i]], 1.0f);
    atomicAdd(&deg[dst[i]], 1.0f);
  }
}

__global__ void dinv_kernel(const float* __restrict__ deg, float* __restrict__ dinv,
                            float* __restrict__ dis, int n) {
  int i = blockIdx.x * blockDim.x + threadIdx.x;
  if (i >= n) return;
  float d  = deg[i];
  float dv = d > 0.0f ? 1.0f / d : 0.0f;
  dinv[i] = dv;
  dis[i]  = sqrtf(dv);
}

// ---------------- fused GEMM: f = relu(x@W1+b1)@W2+b2 -----------------------
// 128 threads = 4 waves; each wave owns a 16-row tile.
__global__ __launch_bounds__(128)
void gemm_fused_kernel(const float* __restrict__ x, const float* __restrict__ b1,
                       const float* __restrict__ b2, const half_t* __restrict__ W1f,
                       const half_t* __restrict__ W2f, float* __restrict__ f, int n) {
  __shared__ half_t hst[4][16 * N_HID];   // per-wave 16x64 f16 h tile (8 KB)

  const int lane = threadIdx.x & 31;
  const int wave = threadIdx.x >> 5;
  const int g    = lane >> 4;             // lane half
  const int nn   = lane & 15;             // M (A) / N (B,C,D) index
  const long long row0 = ((long long)blockIdx.x * 4 + wave) * 16;

  long long ar = row0 + nn;               // clamped A row (tail-safe)
  if (ar > (long long)n - 1) ar = (long long)n - 1;
  const float* xrow = x + ar * (long long)N_IN;

  V8F acc[4];
  #pragma unroll
  for (int t = 0; t < 4; ++t)
    #pragma unroll
    for (int r = 0; r < 8; ++r) acc[t].f[r] = 0.0f;

  // GEMM1: K = 512 in 16 steps of 32
  for (int kb = 0; kb < N_IN / 32; ++kb) {
    V16H a;
    #pragma unroll
    for (int p = 0; p < 8; ++p) {
      const int e = p * 2;
      const int K = e + g * 16;
      const float2 xv = *(const float2*)(xrow + kb * 32 + K);
      a.h[e]     = (half_t)xv.x;
      a.h[e + 1] = (half_t)xv.y;
    }
    #pragma unroll
    for (int t = 0; t < 4; ++t) {
      V16H b;
      b.v = *(const v16h*)(W1f + (((size_t)kb * 4 + t) * 32 + lane) * 16);
      acc[t].v = __builtin_amdgcn_wmma_f32_16x16x32_f16(
          false, a.v, false, b.v, (short)0, acc[t].v, false, false);
    }
  }

  // bias + relu -> LDS f16 tile (C/D layout: row = r + 8g, col = 16t + nn)
  half_t* hs = hst[wave];
  #pragma unroll
  for (int t = 0; t < 4; ++t)
    #pragma unroll
    for (int r = 0; r < 8; ++r) {
      const int col = t * 16 + nn;
      const int row = r + g * 8;
      float v = acc[t].f[r] + b1[col];
      hs[row * N_HID + col] = (half_t)(v > 0.0f ? v : 0.0f);
    }
  __syncthreads();

  // GEMM2: K = 64 in 2 steps of 32
  V8F fo[2];
  #pragma unroll
  for (int t = 0; t < 2; ++t)
    #pragma unroll
    for (int r = 0; r < 8; ++r) fo[t].f[r] = 0.0f;

  #pragma unroll
  for (int ks = 0; ks < 2; ++ks) {
    V16H a;
    #pragma unroll
    for (int p = 0; p < 8; ++p) {
      const int e   = p * 2;
      const int K   = e + g * 16;
      const int col = ks * 32 + K;
      a.h[e]     = hs[nn * N_HID + col];
      a.h[e + 1] = hs[nn * N_HID + col + 1];
    }
    #pragma unroll
    for (int t = 0; t < 2; ++t) {
      V16H b;
      b.v = *(const v16h*)(W2f + (((size_t)ks * 2 + t) * 32 + lane) * 16);
      fo[t].v = __builtin_amdgcn_wmma_f32_16x16x32_f16(
          false, a.v, false, b.v, (short)0, fo[t].v, false, false);
    }
  }

  #pragma unroll
  for (int t = 0; t < 2; ++t)
    #pragma unroll
    for (int r = 0; r < 8; ++r) {
      const long long row = row0 + r + g * 8;
      const int col = t * 16 + nn;
      if (row < n) f[row * N_OUT + col] = fo[t].f[r] + b2[col];
    }
}

// ---------------- propagation ----------------------------------------------
__global__ __launch_bounds__(256)
void gs_kernel(const float* __restrict__ in, const float* __restrict__ dis,
               float* __restrict__ gs, int n) {
  long long i = (long long)blockIdx.x * blockDim.x + threadIdx.x;
  if (i < (long long)n * N_OUT) gs[i] = in[i] * dis[i >> 5];
}

// one wave per edge; lane = channel (N_OUT == 32 == wave32)
__global__ __launch_bounds__(256)
void edge_kernel(const int* __restrict__ src, const int* __restrict__ dst,
                 const float* __restrict__ gs, float* __restrict__ s,
                 float* __restrict__ agg, int n_edges) {
  const int lane = threadIdx.x & 31;
  const long long e = ((long long)blockIdx.x * blockDim.x + threadIdx.x) >> 5;
  if (e >= n_edges) return;   // wave-uniform
  const int a = src[e];
  const int b = dst[e];
  const float ga = gs[(size_t)a * N_OUT + lane];
  const float gb = gs[(size_t)b * N_OUT + lane];
  const float d  = ga - gb;
  float t = d * d;
  #pragma unroll
  for (int off = 16; off > 0; off >>= 1) t += __shfl_xor(t, off, 32);
  const float w = powf(t + EPS_C, PEXP_C);
  if (lane == 0) {
    atomicAdd(&s[a], w);
    atomicAdd(&s[b], w);
  }
  atomicAdd(&agg[(size_t)a * N_OUT + lane], w * gb);
  atomicAdd(&agg[(size_t)b * N_OUT + lane], w * ga);
}

__global__ __launch_bounds__(256)
void update_kernel(const float* __restrict__ agg, const float* __restrict__ s,
                   const float* __restrict__ dinv, const float* __restrict__ dis,
                   const float* __restrict__ f, float* __restrict__ out, int n) {
  long long i = (long long)blockIdx.x * blockDim.x + threadIdx.x;
  if (i >= (long long)n * N_OUT) return;
  const long long row = i >> 5;
  const float alpha = 1.0f / (s[row] * dinv[row] + MU_C);
  const float beta  = MU_C * alpha;
  out[i] = alpha * (dis[row] * agg[i]) + beta * f[i];
}

// ---------------- log_softmax (wave per row) --------------------------------
__global__ __launch_bounds__(256)
void logsoftmax_kernel(const float* __restrict__ in, float* __restrict__ out, int n) {
  const int lane = threadIdx.x & 31;
  const long long row = ((long long)blockIdx.x * blockDim.x + threadIdx.x) >> 5;
  if (row >= n) return;   // wave-uniform
  const float v = in[row * N_OUT + lane];
  float mx = v;
  #pragma unroll
  for (int off = 16; off > 0; off >>= 1) mx = fmaxf(mx, __shfl_xor(mx, off, 32));
  float sum = expf(v - mx);
  #pragma unroll
  for (int off = 16; off > 0; off >>= 1) sum += __shfl_xor(sum, off, 32);
  out[row * N_OUT + lane] = v - mx - logf(sum);
}

// ---------------------------------------------------------------------------
extern "C" void kernel_launch(void* const* d_in, const int* in_sizes, int n_in,
                              void* d_out, int out_size, void* d_ws, size_t ws_size,
                              hipStream_t stream) {
  const float* x  = (const float*)d_in[0];
  const int*   ei = (const int*)  d_in[1];
  const float* W1 = (const float*)d_in[2];
  const float* b1 = (const float*)d_in[3];
  const float* W2 = (const float*)d_in[4];
  const float* b2 = (const float*)d_in[5];

  const int n_edges = in_sizes[1] / 2;
  const int n       = in_sizes[0] / N_IN;
  const int* src = ei;
  const int* dst = ei + n_edges;

  // workspace carve-out (256B aligned)
  char* p = (char*)d_ws;
  auto carve = [&](size_t bytes) -> char* {
    char* r = p;
    p += (bytes + 255) & ~(size_t)255;
    return r;
  };
  half_t* W1f  = (half_t*)carve((size_t)N_IN * N_HID * sizeof(half_t));
  half_t* W2f  = (half_t*)carve((size_t)N_HID * N_OUT * sizeof(half_t));
  float*  deg  = (float*) carve((size_t)n * sizeof(float));
  float*  dinv = (float*) carve((size_t)n * sizeof(float));
  float*  dis  = (float*) carve((size_t)n * sizeof(float));
  float*  fbuf = (float*) carve((size_t)n * N_OUT * sizeof(float));
  float*  obuf = (float*) carve((size_t)n * N_OUT * sizeof(float));
  float*  gs   = (float*) carve((size_t)n * N_OUT * sizeof(float));
  float*  agg  = (float*) carve((size_t)n * N_OUT * sizeof(float));
  float*  sbuf = (float*) carve((size_t)n * sizeof(float));

  const long long nc = (long long)n * N_OUT;
  const int B = 256;
  auto gsz = [](long long t, int b) { return (unsigned)((t + b - 1) / b); };

  // weight fragment tables
  convert_w1_kernel<<<gsz(N_IN * N_HID, B), B, 0, stream>>>(W1, W1f);
  convert_w2_kernel<<<gsz(N_HID * N_OUT, B), B, 0, stream>>>(W2, W2f);

  // degrees
  zero_kernel<<<gsz(n, B), B, 0, stream>>>(deg, n);
  deg_kernel<<<gsz(n_edges, B), B, 0, stream>>>(src, dst, deg, n_edges);
  dinv_kernel<<<gsz(n, B), B, 0, stream>>>(deg, dinv, dis, n);

  // fused WMMA MLP: fbuf = relu(x@W1+b1)@W2+b2
  const long long tiles = (n + 15) / 16;
  gemm_fused_kernel<<<gsz(tiles, 4), 128, 0, stream>>>(x, b1, b2, W1f, W2f, fbuf, n);

  // K = 2 propagation rounds
  const float* cur = fbuf;
  for (int it = 0; it < 2; ++it) {
    gs_kernel<<<gsz(nc, B), B, 0, stream>>>(cur, dis, gs, n);
    zero_kernel<<<gsz(n, B), B, 0, stream>>>(sbuf, n);
    zero_kernel<<<gsz(nc, B), B, 0, stream>>>(agg, nc);
    edge_kernel<<<gsz((long long)n_edges * 32, B), B, 0, stream>>>(src, dst, gs, sbuf, agg, n_edges);
    update_kernel<<<gsz(nc, B), B, 0, stream>>>(agg, sbuf, dinv, dis, fbuf, obuf, n);
    cur = obuf;
  }

  logsoftmax_kernel<<<gsz(nc, B), B, 0, stream>>>(cur, (float*)d_out, n);
}
